// GraphEncoder_5162550689834
// MI455X (gfx1250) — matrix operand order repaired and checked by reading, not compiled, not run on hardware
//
#include <hip/hip_runtime.h>

typedef __attribute__((ext_vector_type(16))) _Float16 v16h;
typedef __attribute__((ext_vector_type(8)))  _Float16 v8h;
typedef __attribute__((ext_vector_type(8)))  float    v8f;

#define F    128
#define KCAT 256
#define EPS  1e-5f

// ---------------- utility ----------------
__global__ void zero_f32_kernel(float* __restrict__ p, long n) {
  long i = blockIdx.x * (long)blockDim.x + threadIdx.x;
  long stride = (long)gridDim.x * blockDim.x;
  for (; i < n; i += stride) p[i] = 0.0f;
}

// ---------------- degree ----------------
__global__ void deg_kernel(const int* __restrict__ dst, float* __restrict__ deg, int E) {
  int e = blockIdx.x * blockDim.x + threadIdx.x;
  if (e < E) atomicAdd(&deg[dst[e]], 1.0f);
}

// ---------------- edge scatter: msg[dst] += h[src] (L2-resident atomics) ----------------
__global__ void scatter_kernel(const float* __restrict__ hin, const int* __restrict__ src,
                               const int* __restrict__ dst, float* __restrict__ msg, int E) {
  long tid = blockIdx.x * (long)blockDim.x + threadIdx.x;
  long tot = (long)E * 32;  // 32 threads per edge, 4 floats each
  if (tid >= tot) return;
  int e  = (int)(tid >> 5);
  int f4 = (int)(tid & 31) << 2;
  int s = src[e], d = dst[e];
  const float4 v = *(const float4*)(hin + (long)s * F + f4);
  float* o = msg + (long)d * F + f4;
  atomicAdd(o + 0, v.x);
  atomicAdd(o + 1, v.y);
  atomicAdd(o + 2, v.z);
  atomicAdd(o + 3, v.w);
}

// ---------------- build f16 A matrix: [ msg/deg | h ]  (N x 256) ----------------
__global__ void acat_kernel(const float* __restrict__ msg, const float* __restrict__ deg,
                            const float* __restrict__ hin, _Float16* __restrict__ acat, int N) {
  long tid = blockIdx.x * (long)blockDim.x + threadIdx.x;
  if (tid >= (long)N * F) return;
  int i = (int)(tid / F), f = (int)(tid % F);
  float dg = deg[i];
  dg = dg < 1.0f ? 1.0f : dg;
  acat[(long)i * KCAT + f]     = (_Float16)(msg[tid] / dg);
  acat[(long)i * KCAT + F + f] = (_Float16)hin[tid];
}

// ---------------- weights: WcatT[n][k] = f16( k<128 ? w_l[k][n] : w_r[k-128][n] ) ----------------
__global__ void wprep_kernel(const float* __restrict__ wl, const float* __restrict__ wr,
                             _Float16* __restrict__ wcatT) {
  int tid = blockIdx.x * blockDim.x + threadIdx.x;
  if (tid >= F * KCAT) return;
  int n = tid / KCAT, k = tid % KCAT;
  float v = (k < F) ? wl[k * F + n] : wr[(k - F) * F + n];
  wcatT[n * KCAT + k] = (_Float16)v;
}

// ---------------- fused GEMM: out = Acat(f16) @ Wcat(f16) + bias, + per-col BN sums ----------------
// Block = 256 threads = 8 waves. Block covers 16 rows x 128 cols; wave w -> cols [16w,16w+16).
__global__ __launch_bounds__(256)
void gemm_wmma_kernel(const _Float16* __restrict__ acat, const _Float16* __restrict__ wcatT,
                      const float* __restrict__ bias, float* __restrict__ out,
                      float* __restrict__ colsum, float* __restrict__ colsumsq, int N) {
  int wave = threadIdx.x >> 5;
  int lane = threadIdx.x & 31;
  int m0 = blockIdx.x * 16;
  int n0 = wave * 16;
  int mrow = lane & 15;
  int hi   = lane >> 4;

  int arowidx = m0 + mrow;
  if (arowidx >= N) arowidx = N - 1;  // clamp loads; stores guarded below

  // A fragment (16x32 f16): lane half 'hi' owns K groups {hi*8 + 0..7, hi*8 + 16..23}
  const _Float16* arow = acat + (long)arowidx * KCAT + hi * 8;
  // B fragment (32x16 f16): lane half 'hi' owns K groups {hi*8 + 0..7, hi*8 + 16..23} of column n0+(lane&15)
  const _Float16* brow = wcatT + (long)(n0 + (lane & 15)) * KCAT + hi * 8;

  v8f c = {};
#pragma unroll
  for (int kk = 0; kk < 8; ++kk) {
    int kb = kk * 32;
    union { v16h v; v8h h[2]; } a, b;
    a.h[0] = *(const v8h*)(arow + kb);        // K = kb + hi*8 + (0..7)
    a.h[1] = *(const v8h*)(arow + kb + 16);   // K = kb + hi*8 + 16 + (0..7)
    b.h[0] = *(const v8h*)(brow + kb);        // K = kb + hi*8 + (0..7)
    b.h[1] = *(const v8h*)(brow + kb + 16);   // K = kb + hi*8 + 16 + (0..7)
    c = __builtin_amdgcn_wmma_f32_16x16x32_f16(false, a.v, false, b.v,
                                               (short)0, c, false, false);
  }

  int col = n0 + (lane & 15);
  float bcol = bias[col];
  float s = 0.0f, s2 = 0.0f;
#pragma unroll
  for (int r = 0; r < 8; ++r) {
    int row = m0 + r + hi * 8;  // C layout: VGPR r -> M = r (+8 for lanes 16..31)
    float v = c[r] + bcol;
    if (row < N) {
      out[(long)row * F + col] = v;
      s += v;
      s2 += v * v;
    }
  }
  // lanes L and L+16 share the same output column -> fold, then one atomic per col
  s  += __shfl_xor(s, 16, 32);
  s2 += __shfl_xor(s2, 16, 32);
  if (hi == 0) {
    atomicAdd(&colsum[col], s);
    atomicAdd(&colsumsq[col], s2);
  }
}

// ---------------- batchnorm stats -> affine scale/shift ----------------
__global__ void bnstats_kernel(const float* __restrict__ colsum, const float* __restrict__ colsumsq,
                               const float* __restrict__ g, const float* __restrict__ beta,
                               float* __restrict__ scale, float* __restrict__ shift, int N) {
  int f = threadIdx.x;
  if (f >= F) return;
  float invN = 1.0f / (float)N;
  float mu  = colsum[f] * invN;
  float var = colsumsq[f] * invN - mu * mu;
  float sc  = g[f] * rsqrtf(var + EPS);
  scale[f] = sc;
  shift[f] = beta[f] - mu * sc;
}

// ---------------- batchnorm apply + ReLU ----------------
__global__ void bnrelu_kernel(const float* __restrict__ pre, const float* __restrict__ scale,
                              const float* __restrict__ shift, float* __restrict__ hout, long total) {
  long tid = blockIdx.x * (long)blockDim.x + threadIdx.x;
  if (tid >= total) return;
  int f = (int)(tid % F);
  float v = pre[tid] * scale[f] + shift[f];
  hout[tid] = v > 0.0f ? v : 0.0f;
}

// ---------------- global mean pool ----------------
__global__ void poolacc_kernel(const float* __restrict__ h, const int* __restrict__ batch,
                               float* __restrict__ pooled, float* __restrict__ cnt, int N) {
  int i = blockIdx.x;
  int f = threadIdx.x;
  if (i >= N) return;
  int b = batch[i];
  atomicAdd(&pooled[(long)b * F + f], h[(long)i * F + f]);
  if (f == 0) atomicAdd(&cnt[b], 1.0f);
}

__global__ void pooldiv_kernel(const float* __restrict__ pooled, const float* __restrict__ cnt,
                               float* __restrict__ outp, int B) {
  int tid = blockIdx.x * blockDim.x + threadIdx.x;
  if (tid >= B * F) return;
  int b = tid / F;
  float c = cnt[b];
  c = c < 1.0f ? 1.0f : c;
  outp[tid] = pooled[tid] / c;
}

// ---------------- host launch ----------------
extern "C" void kernel_launch(void* const* d_in, const int* in_sizes, int n_in,
                              void* d_out, int out_size, void* d_ws, size_t ws_size,
                              hipStream_t stream) {
  const float* x     = (const float*)d_in[0];
  const int*   ei    = (const int*)d_in[1];
  const int*   batch = (const int*)d_in[2];
  const float* w_l1  = (const float*)d_in[3];
  const float* b_l1  = (const float*)d_in[4];
  const float* w_r1  = (const float*)d_in[5];
  const float* g1    = (const float*)d_in[6];
  const float* beta1 = (const float*)d_in[7];
  const float* w_l2  = (const float*)d_in[8];
  const float* b_l2  = (const float*)d_in[9];
  const float* w_r2  = (const float*)d_in[10];
  const float* g2    = (const float*)d_in[11];
  const float* beta2 = (const float*)d_in[12];

  const int N = in_sizes[0] / F;
  const int E = in_sizes[1] / 2;
  const int B = out_size / F;
  const int* src = ei;
  const int* dst = ei + E;

  char* ws = (char*)d_ws;
  size_t off = 0;
  auto alloc = [&](size_t bytes) -> void* {
    void* p = ws + off;
    off = (off + bytes + 255) & ~(size_t)255;
    return p;
  };

  float*     msgpre   = (float*)alloc((size_t)N * F * 4);      // msg_sum, reused as GEMM output
  _Float16*  acat     = (_Float16*)alloc((size_t)N * KCAT * 2);
  float*     hA       = (float*)alloc((size_t)N * F * 4);
  float*     hB       = (float*)alloc((size_t)N * F * 4);
  _Float16*  wcatT    = (_Float16*)alloc((size_t)F * KCAT * 2);
  float*     deg      = (float*)alloc((size_t)N * 4);
  float*     colsum   = (float*)alloc(F * 4);
  float*     colsumsq = (float*)alloc(F * 4);
  float*     bnscale  = (float*)alloc(F * 4);
  float*     bnshift  = (float*)alloc(F * 4);
  float*     pooled   = (float*)alloc((size_t)B * F * 4);
  float*     cnt      = (float*)alloc((size_t)B * 4);

  const long nf = (long)N * F;
  const int mtiles = (N + 15) / 16;

  auto run_layer = [&](const float* hin, const float* wl, const float* bl, const float* wr,
                       const float* g, const float* beta, float* hout) {
    zero_f32_kernel<<<2048, 256, 0, stream>>>(msgpre, nf);
    zero_f32_kernel<<<1, 256, 0, stream>>>(colsum, F);
    zero_f32_kernel<<<1, 256, 0, stream>>>(colsumsq, F);
    wprep_kernel<<<(F * KCAT + 255) / 256, 256, 0, stream>>>(wl, wr, wcatT);
    long st = (long)E * 32;
    scatter_kernel<<<(int)((st + 255) / 256), 256, 0, stream>>>(hin, src, dst, msgpre, E);
    acat_kernel<<<(int)((nf + 255) / 256), 256, 0, stream>>>(msgpre, deg, hin, acat, N);
    gemm_wmma_kernel<<<mtiles, 256, 0, stream>>>(acat, wcatT, bl, msgpre, colsum, colsumsq, N);
    bnstats_kernel<<<1, 128, 0, stream>>>(colsum, colsumsq, g, beta, bnscale, bnshift, N);
    bnrelu_kernel<<<(int)((nf + 255) / 256), 256, 0, stream>>>(msgpre, bnscale, bnshift, hout, nf);
  };

  // degrees (shared by both layers)
  zero_f32_kernel<<<256, 256, 0, stream>>>(deg, N);
  deg_kernel<<<(E + 255) / 256, 256, 0, stream>>>(dst, deg, E);

  run_layer(x,  w_l1, b_l1, w_r1, g1, beta1, hA);
  run_layer(hA, w_l2, b_l2, w_r2, g2, beta2, hB);

  // global mean pool
  zero_f32_kernel<<<32, 256, 0, stream>>>(pooled, (long)B * F);
  zero_f32_kernel<<<1, 64, 0, stream>>>(cnt, B);
  poolacc_kernel<<<N, F, 0, stream>>>(hB, batch, pooled, cnt, N);
  pooldiv_kernel<<<(B * F + 255) / 256, 256, 0, stream>>>(pooled, cnt, (float*)d_out, B);
}
